// MambaBlock_8710193676775
// MI455X (gfx1250) — matrix operand order repaired and checked by reading, not compiled, not run on hardware
//
#include <hip/hip_runtime.h>
#include <stddef.h>

// ---------------------------------------------------------------------------
// Types for CDNA5 WMMA
// ---------------------------------------------------------------------------
typedef __bf16 v16bf __attribute__((ext_vector_type(16)));
typedef float  v8f   __attribute__((ext_vector_type(8)));

// Mamba dims (fixed by the reference)
#define D_MODEL 1024
#define D_INNER 2048
#define D_CONV  4
#define D_STATE 16
#define DT_RANK 64
#define B_SZ    2
#define SEQ_L   2048

// Async global->LDS path (gfx1250), guarded so compilation never breaks.
#if defined(__has_builtin)
# if __has_builtin(__builtin_amdgcn_global_load_async_to_lds_b32)
#  define MAMBA_ASYNC_LDS 1
# endif
#endif

typedef __attribute__((address_space(1))) int as1_int;
typedef __attribute__((address_space(3))) int as3_int;

static __device__ __forceinline__ unsigned short f32_to_bf16(float f) {
    unsigned int u = __builtin_bit_cast(unsigned int, f);
    unsigned int r = u + 0x7FFFu + ((u >> 16) & 1u);   // round-to-nearest-even
    return (unsigned short)(r >> 16);
}
static __device__ __forceinline__ unsigned int pack_bf16x2(float a, float b) {
    return (unsigned int)f32_to_bf16(a) | ((unsigned int)f32_to_bf16(b) << 16);
}
static __device__ __forceinline__ __bf16 us_as_bf(unsigned short u) {
    return __builtin_bit_cast(__bf16, u);
}
static __device__ __forceinline__ float softplusf(float x) {
    return (x > 20.f) ? x : log1pf(__expf(x));
}

// ---------------------------------------------------------------------------
// Generic WMMA GEMM:  C[M,N] (f32) = A[M,K] (f32, row-major, lda)
//                                  @ B[K,N] (f32, row-major, ldb)
// f32 operands converted to bf16 while staging into LDS (packed pairs,
// b32 LDS stores); accumulation f32 via v_wmma_f32_16x16x32_bf16.
// Both A and B(Transposed) LDS tiles have 80-byte (16B-aligned) row pitch so
// fragment reads lower to ds_load_b128.
// Block tile: 64(M) x 256(N), K step 32. 256 threads = 8 waves (2 M x 4 N),
// each wave computes a 32x64 sub-tile (2x4 WMMA accumulators): 8 WMMA per
// K-step per wave, each A fragment reused across 4 B fragments.
// epilogue: 0 = none, 1 = softplus(acc + bias[col])
// ---------------------------------------------------------------------------
#define BM 64
#define BN 256
#define BK 32
#define LDS_PITCH_U32 20   // 16 used + pad; 80 B rows keep 16-byte alignment

__launch_bounds__(256)
__global__ void mamba_gemm_wmma_bf16(const float* __restrict__ A, int lda,
                                     const float* __restrict__ B, int ldb,
                                     float* __restrict__ C, int ldc,
                                     int M, int N, int K,
                                     int epilogue, const float* __restrict__ bias) {
    __shared__ unsigned int As32[BM][LDS_PITCH_U32];   // [row][k-pair]   (~5 KB)
    __shared__ unsigned int Bs32[BN][LDS_PITCH_U32];   // [col][k-pair]  (~20 KB)

    const unsigned short* As16 = (const unsigned short*)&As32[0][0];
    const unsigned short* Bs16 = (const unsigned short*)&Bs32[0][0];

    const int tid    = threadIdx.x;
    const int lane   = tid & 31;
    const int wave   = tid >> 5;
    const int blockM = blockIdx.y * BM;
    const int blockN = blockIdx.x * BN;
    const int waveM  = (wave & 1) * 32;      // 2 waves along M
    const int waveN  = (wave >> 1) * 64;     // 4 waves along N, 64 cols each

    v8f acc[2][4];
#pragma unroll
    for (int m = 0; m < 2; ++m)
#pragma unroll
        for (int n = 0; n < 4; ++n)
#pragma unroll
            for (int r = 0; r < 8; ++r) acc[m][n][r] = 0.f;

    const int arow = lane & 15;
    const int kbA  = (lane >> 4) * 8;    // A: ushort K base per lane half
    const int bcol = lane & 15;
    const int kbB  = (lane >> 4) * 8;    // B: ushort K base per lane half

    for (int k0 = 0; k0 < K; k0 += BK) {
        // --- stage A tile (64 rows x 16 bf16-pairs): 1024 u32, 4 per thread.
        // Branch-free row clamp: clamped rows feed only masked output rows.
#pragma unroll
        for (int j = 0; j < 4; ++j) {
            int idx = j * 256 + tid;
            int r = idx >> 4, kp = idx & 15;
            int gr = blockM + r;
            if (gr > M - 1) gr = M - 1;
            const float* ap = &A[(size_t)gr * lda + (k0 + 2 * kp)];
            As32[r][kp] = pack_bf16x2(ap[0], ap[1]);
        }
        // --- stage B tile transposed (256 cols x 16 bf16-pairs): 4096 u32,
        // 16 per thread. Branch-free col clamp + select-zero.
#pragma unroll
        for (int j = 0; j < 16; ++j) {
            int idx = j * 256 + tid;
            int c = idx >> 4, kp = idx & 15;
            int gc = blockN + c;
            int gcc = (gc < N) ? gc : (N - 1);
            int kr = k0 + 2 * kp;
            float f0 = B[(size_t)kr * ldb + gcc];
            float f1 = B[(size_t)(kr + 1) * ldb + gcc];
            unsigned int p = pack_bf16x2(f0, f1);
            if (gc >= N) p = 0u;
            Bs32[c][kp] = p;
        }
        __syncthreads();

        // --- build fragments per ISA VGPR layouts (contiguous 16B runs) ---
        v16bf afrag[2], bfrag[4];
#pragma unroll
        for (int m = 0; m < 2; ++m) {
            int rr = waveM + m * 16 + arow;
            int base = rr * (2 * LDS_PITCH_U32);   // ushort index of row start
#pragma unroll
            for (int e = 0; e < 16; ++e) {
                int kk = base + ((e >= 8) ? 16 : 0) + kbA + (e & 7);
                afrag[m][e] = us_as_bf(As16[kk]);
            }
        }
#pragma unroll
        for (int n = 0; n < 4; ++n) {
            int cc = waveN + n * 16 + bcol;
            int base = cc * (2 * LDS_PITCH_U32) + kbB;
#pragma unroll
            for (int e = 0; e < 16; ++e) {
                bfrag[n][e] = us_as_bf(Bs16[base + ((e >= 8) ? 16 : 0) + (e & 7)]);
            }
        }

#pragma unroll
        for (int m = 0; m < 2; ++m)
#pragma unroll
            for (int n = 0; n < 4; ++n)
                acc[m][n] = __builtin_amdgcn_wmma_f32_16x16x32_bf16(
                    /*neg_a=*/false, afrag[m], /*neg_b=*/false, bfrag[n],
                    /*c_mod=*/(short)0, acc[m][n],
                    /*reuse_a=*/false, /*reuse_b=*/false);
        __syncthreads();
    }

    // --- store C (f32 16x16 C/D layout: VGPR r -> row r + 8*(lane>=16)) ---
    const int rofs = (lane >> 4) * 8;
#pragma unroll
    for (int m = 0; m < 2; ++m) {
#pragma unroll
        for (int n = 0; n < 4; ++n) {
            int gc = blockN + waveN + n * 16 + (lane & 15);
#pragma unroll
            for (int r = 0; r < 8; ++r) {
                int gr = blockM + waveM + m * 16 + rofs + r;
                if (gr < M && gc < N) {
                    float v = acc[m][n][r];
                    if (epilogue == 1) v = softplusf(v + bias[gc]);
                    C[gr * ldc + gc] = v;
                }
            }
        }
    }
}

// ---------------------------------------------------------------------------
// Causal depthwise conv1d (kernel 4) + bias + SiLU.
// xi = xr[..., 0:D_INNER];  xc[b,l,d] = silu(sum_j w[d,j]*xi[b,l-3+j,d] + b[d])
// ---------------------------------------------------------------------------
__launch_bounds__(256)
__global__ void mamba_conv_silu(const float* __restrict__ xr,
                                const float* __restrict__ w,
                                const float* __restrict__ bias,
                                float* __restrict__ xc) {
    size_t i = (size_t)blockIdx.x * 256 + threadIdx.x;   // over B*L*D_INNER
    int d = (int)(i % D_INNER);
    size_t t = i / D_INNER;
    int l = (int)(t % SEQ_L);
    int b = (int)(t / SEQ_L);
    float acc = bias[d];
#pragma unroll
    for (int j = 0; j < D_CONV; ++j) {
        int lj = l + j - (D_CONV - 1);
        if (lj >= 0)
            acc += w[d * D_CONV + j] *
                   xr[((size_t)(b * SEQ_L + lj)) * (2 * D_INNER) + d];
    }
    xc[i] = acc / (1.f + __expf(-acc));   // SiLU
}

// ---------------------------------------------------------------------------
// Selective scan, fused with u*D skip and SiLU(res) gating.
// One thread per (b, d): 16 SSM states in registers; B/C vectors (shared by
// all d at a given l) staged through LDS in 64-step chunks, using
// GLOBAL_LOAD_ASYNC_TO_LDS_B32 when the toolchain exposes it.
// ---------------------------------------------------------------------------
__launch_bounds__(256)
__global__ void mamba_scan_gate(const float* __restrict__ delta,
                                const float* __restrict__ xc,
                                const float* __restrict__ xr,     // res = xr[..., D_INNER:]
                                const float* __restrict__ x_dbl,  // [.., 96]: B at 64, C at 80
                                const float* __restrict__ A_log,
                                const float* __restrict__ Dvec,
                                float* __restrict__ y) {
    const int d = blockIdx.x * 256 + threadIdx.x;   // 0 .. D_INNER-1
    const int b = blockIdx.y;

    __shared__ float BC[64][32];   // [l_local][n<16 : B, n>=16 : C]

    float An[D_STATE], s[D_STATE];
#pragma unroll
    for (int n = 0; n < D_STATE; ++n) {
        An[n] = -__expf(A_log[d * D_STATE + n]);
        s[n] = 0.f;
    }
    const float Dd = Dvec[d];

    for (int l0 = 0; l0 < SEQ_L; l0 += 64) {
        __syncthreads();   // previous chunk fully consumed
#ifdef MAMBA_ASYNC_LDS
        for (int i = threadIdx.x; i < 64 * 32; i += 256) {
            int ll = i >> 5, n = i & 31;
            const float* src = &x_dbl[((size_t)(b * SEQ_L + l0 + ll)) *
                                          (DT_RANK + 2 * D_STATE) + DT_RANK + n];
            __builtin_amdgcn_global_load_async_to_lds_b32(
                (as1_int*)(int*)(float*)src,
                (as3_int*)(int*)&BC[ll][n],
                0, 0);
        }
# if defined(__has_builtin) && __has_builtin(__builtin_amdgcn_s_wait_asynccnt)
        __builtin_amdgcn_s_wait_asynccnt(0);
# else
        asm volatile("s_wait_asynccnt 0x0" ::: "memory");
# endif
#else
        for (int i = threadIdx.x; i < 64 * 32; i += 256) {
            int ll = i >> 5, n = i & 31;
            BC[ll][n] = x_dbl[((size_t)(b * SEQ_L + l0 + ll)) *
                                  (DT_RANK + 2 * D_STATE) + DT_RANK + n];
        }
#endif
        __syncthreads();

        for (int lt = 0; lt < 64; ++lt) {
            size_t base = (size_t)(b * SEQ_L + l0 + lt);
            float dv  = delta[base * D_INNER + d];
            float u   = xc[base * D_INNER + d];
            float res = xr[base * (2 * D_INNER) + D_INNER + d];
            float yv = 0.f;
#pragma unroll
            for (int n = 0; n < D_STATE; ++n) {
                float dA  = __expf(dv * An[n]);
                float dBu = dv * BC[lt][n] * u;
                s[n] = s[n] * dA + dBu;
                yv += s[n] * BC[lt][16 + n];
            }
            yv += u * Dd;
            float g = res / (1.f + __expf(-res));  // silu(res)
            y[base * D_INNER + d] = yv * g;
        }
    }
}

// ---------------------------------------------------------------------------
// Host-side launcher
// ---------------------------------------------------------------------------
extern "C" void kernel_launch(void* const* d_in, const int* in_sizes, int n_in,
                              void* d_out, int out_size, void* d_ws, size_t ws_size,
                              hipStream_t stream) {
    (void)in_sizes; (void)n_in; (void)out_size; (void)ws_size;

    const float* x         = (const float*)d_in[0];  // (B,L,1024)
    const float* in_proj_w = (const float*)d_in[1];  // (1024,4096)
    const float* conv_w    = (const float*)d_in[2];  // (2048,1,4)
    const float* conv_b    = (const float*)d_in[3];  // (2048,)
    const float* x_proj_w  = (const float*)d_in[4];  // (2048,96)
    const float* dt_proj_w = (const float*)d_in[5];  // (64,2048)
    const float* dt_proj_b = (const float*)d_in[6];  // (2048,)
    const float* A_log     = (const float*)d_in[7];  // (2048,16)
    const float* Dvec      = (const float*)d_in[8];  // (2048,)
    const float* out_proj_w= (const float*)d_in[9];  // (2048,1024)
    float* out = (float*)d_out;                      // (B,L,1024)

    const int M = B_SZ * SEQ_L;                      // 4096 rows
    char* ws = (char*)d_ws;
    size_t off = 0;
    float* xr    = (float*)(ws + off); off += (size_t)M * (2 * D_INNER) * 4;        // 64 MB
    float* xc    = (float*)(ws + off); off += (size_t)M * D_INNER * 4;              // 32 MB
    float* x_dbl = (float*)(ws + off); off += (size_t)M * (DT_RANK + 2*D_STATE)*4;  // 1.5 MB
    float* delta = (float*)(ws + off); off += (size_t)M * D_INNER * 4;              // 32 MB
    float* y     = (float*)(ws + off); off += (size_t)M * D_INNER * 4;              // 32 MB

    dim3 blk(256);

    // 1) in_proj: xr[4096,4096] = x[4096,1024] @ in_proj_w[1024,4096]
    {
        dim3 grid((2 * D_INNER + BN - 1) / BN, (M + BM - 1) / BM);
        mamba_gemm_wmma_bf16<<<grid, blk, 0, stream>>>(
            x, D_MODEL, in_proj_w, 2 * D_INNER, xr, 2 * D_INNER,
            M, 2 * D_INNER, D_MODEL, 0, nullptr);
    }
    // 2) causal depthwise conv + SiLU -> xc
    {
        size_t total = (size_t)M * D_INNER;
        mamba_conv_silu<<<dim3((unsigned)(total / 256)), blk, 0, stream>>>(
            xr, conv_w, conv_b, xc);
    }
    // 3) x_proj: x_dbl[4096,96] = xc @ x_proj_w[2048,96]
    {
        dim3 grid((DT_RANK + 2 * D_STATE + BN - 1) / BN, (M + BM - 1) / BM);
        mamba_gemm_wmma_bf16<<<grid, blk, 0, stream>>>(
            xc, D_INNER, x_proj_w, DT_RANK + 2 * D_STATE, x_dbl,
            DT_RANK + 2 * D_STATE, M, DT_RANK + 2 * D_STATE, D_INNER, 0, nullptr);
    }
    // 4) delta = softplus(dt @ dt_proj_w + dt_proj_b); dt = x_dbl[:, :64] (lda=96)
    {
        dim3 grid((D_INNER + BN - 1) / BN, (M + BM - 1) / BM);
        mamba_gemm_wmma_bf16<<<grid, blk, 0, stream>>>(
            x_dbl, DT_RANK + 2 * D_STATE, dt_proj_w, D_INNER, delta, D_INNER,
            M, D_INNER, DT_RANK, 1, dt_proj_b);
    }
    // 5) selective scan + u*D + SiLU(res) gate -> y
    {
        dim3 grid(D_INNER / 256, B_SZ);
        mamba_scan_gate<<<grid, blk, 0, stream>>>(
            delta, xc, xr, x_dbl, A_log, Dvec, y);
    }
    // 6) out_proj: out[4096,1024] = y @ out_proj_w[2048,1024]
    {
        dim3 grid((D_MODEL + BN - 1) / BN, (M + BM - 1) / BM);
        mamba_gemm_wmma_bf16<<<grid, blk, 0, stream>>>(
            y, D_INNER, out_proj_w, D_MODEL, out, D_MODEL,
            M, D_MODEL, D_INNER, 0, nullptr);
    }
}
